// DecoderBertModel_9363028705312
// MI455X (gfx1250) — matrix operand order repaired and checked by reading, not compiled, not run on hardware
//
#include <hip/hip_runtime.h>
#include <math.h>

// ---------------- model constants ----------------
#define kH    768
#define kNH   12
#define kDH   64
#define kFF   3072
#define kL    6
#define kV    30522
#define kVpad 30592   // kV rounded up to 128 (tile N), pad rows zeroed
#define kB    4
#define kS    512
#define kSE   512
#define kMR   (kB * kS)    // 2048 rows of decoder activations
#define kBN   (kB * kNH)   // 48 attention batches
#define kVTD  128          // padded per-head V^T rows (>= kDH, tile-N aligned)

typedef __attribute__((ext_vector_type(16))) _Float16 v16h;
typedef __attribute__((ext_vector_type(8)))  _Float16 v8h;
typedef __attribute__((ext_vector_type(8)))  float    v8f;

// ---------------- small helpers ----------------
__device__ __forceinline__ v16h cat8(v8h lo, v8h hi) {
  v16h r;
#pragma unroll
  for (int i = 0; i < 8; ++i) { r[i] = lo[i]; r[i + 8] = hi[i]; }
  return r;
}

__device__ __forceinline__ float gelu_f(float x) {
  return 0.5f * x * (1.0f + erff(x * 0.70710678118654752440f));
}

__device__ __forceinline__ float blk_sum256(float v) {
  __shared__ float sd[256];
  int t = threadIdx.x;
  sd[t] = v; __syncthreads();
  for (int s = 128; s > 0; s >>= 1) { if (t < s) sd[t] += sd[t + s]; __syncthreads(); }
  float r = sd[0]; __syncthreads();
  return r;
}

__device__ __forceinline__ float blk_max256(float v) {
  __shared__ float sm[256];
  int t = threadIdx.x;
  sm[t] = v; __syncthreads();
  for (int s = 128; s > 0; s >>= 1) { if (t < s) sm[t] = fmaxf(sm[t], sm[t + s]); __syncthreads(); }
  float r = sm[0]; __syncthreads();
  return r;
}

// ---------------- fp32 -> fp16 convert / f16 zero-fill ----------------
__global__ void convert_f16_kernel(const float* __restrict__ in,
                                   _Float16* __restrict__ out, long long n) {
  long long i = (long long)blockIdx.x * blockDim.x + threadIdx.x;
  if (i < n) out[i] = (_Float16)in[i];
}

__global__ void zero_f16_kernel(_Float16* __restrict__ out, long long n) {
  long long i = (long long)blockIdx.x * blockDim.x + threadIdx.x;
  if (i < n) out[i] = (_Float16)0.0f;
}

// ---------------- generic batched WMMA GEMM ----------------
// C[M,N] = A[M,K] * B[N,K]^T (+bias) (+gelu); per-batch-z pointer offsets.
// Block: 256 threads = 8 waves (4 M x 2 N); block tile 128x128; wave tile 32x64.
// B operand must have >= ceil(N,128) valid (padded) rows: loads are UNGUARDED.
// STGUARD instantiation bounds-checks only the store epilogue (ctx / vocab).
template <bool STGUARD>
__global__ __launch_bounds__(256) void gemm_wmma_kernel(
    const _Float16* __restrict__ A, const _Float16* __restrict__ Bw,
    const float* __restrict__ bias,
    float* __restrict__ Cf, _Float16* __restrict__ Ch,
    int M, int N, int K, int lda, int ldb, int ldc,
    int zDiv,
    long long aSb, long long aSh, long long bSb, long long bSh,
    long long cSb, long long cSh, int act) {
  int z  = blockIdx.z;
  int zb = z / zDiv, zh = z % zDiv;
  A  += (size_t)zb * aSb + (size_t)zh * aSh;
  Bw += (size_t)zb * bSb + (size_t)zh * bSh;
  size_t cOff = (size_t)zb * cSb + (size_t)zh * cSh;

  const int lane   = threadIdx.x & 31;
  const int wave   = threadIdx.x >> 5;
  const int wm     = wave >> 1;   // 0..3
  const int wn     = wave & 1;    // 0..1
  const int lane16 = lane & 15;
  const int hi     = lane >> 4;

  const int m0 = blockIdx.y * 128 + wm * 32;
  const int n0 = blockIdx.x * 128 + wn * 64;

  v8f acc[2][4];
#pragma unroll
  for (int i = 0; i < 2; ++i)
#pragma unroll
    for (int j = 0; j < 4; ++j)
#pragma unroll
      for (int r = 0; r < 8; ++r) acc[i][j][r] = 0.0f;

  for (int k = 0; k < K; k += 32) {
    v16h af[2], bf[4];
#pragma unroll
    for (int i = 0; i < 2; ++i) {
      // gfx1250 16-bit A 16x32 layout: lane-lo K={0..7,16..23}, lane-hi +8
      const _Float16* ap = A + (size_t)(m0 + i * 16 + lane16) * lda + k + hi * 8;
      v8h lo  = *(const v8h*)ap;
      v8h hi8 = *(const v8h*)(ap + 16);
      af[i] = cat8(lo, hi8);
    }
#pragma unroll
    for (int j = 0; j < 4; ++j) {
      // B 32x16 layout: lane = column, K={0..7,16..23} per lane (hi lanes +8)
      const _Float16* bp = Bw + (size_t)(n0 + j * 16 + lane16) * ldb + k + hi * 8;
      v8h lo  = *(const v8h*)bp;
      v8h hi8 = *(const v8h*)(bp + 16);
      bf[j] = cat8(lo, hi8);
    }
    // unconditional speculative prefetch (global_prefetch_b8, WGP-temporal);
    // out-of-range speculative prefetches are silently dropped by hardware.
    __builtin_prefetch(A  + (size_t)(m0 + lane16) * lda + k + 64, 0, 3);
    __builtin_prefetch(Bw + (size_t)(n0 + lane16) * ldb + k + 64, 0, 3);
#pragma unroll
    for (int i = 0; i < 2; ++i)
#pragma unroll
      for (int j = 0; j < 4; ++j)
        acc[i][j] = __builtin_amdgcn_wmma_f32_16x16x32_f16(
            false, af[i], false, bf[j], (short)0, acc[i][j], false, false);
  }

  // epilogue: bias (+gelu), dual f32/f16 store. C layout: row = r + 8*hi.
#pragma unroll
  for (int i = 0; i < 2; ++i) {
#pragma unroll
    for (int j = 0; j < 4; ++j) {
      int n = n0 + j * 16 + lane16;
      if (STGUARD && n >= N) continue;
      float bv = bias ? bias[n] : 0.0f;
#pragma unroll
      for (int r = 0; r < 8; ++r) {
        int m = m0 + i * 16 + hi * 8 + r;
        float v = acc[i][j][r] + bv;
        if (act) v = gelu_f(v);
        size_t idx = cOff + (size_t)m * ldc + n;
        if (Cf) Cf[idx] = v;
        if (Ch) Ch[idx] = (_Float16)v;
      }
    }
  }
}

// ---------------- embedding + LayerNorm ----------------
__global__ __launch_bounds__(256) void embed_ln_kernel(
    const int* __restrict__ ids, const float* __restrict__ wemb,
    const float* __restrict__ pemb, const float* __restrict__ g,
    const float* __restrict__ be, float* __restrict__ outF,
    _Float16* __restrict__ outH) {
  int row = blockIdx.x;      // b*kS + s
  int s   = row % kS;
  int id  = ids[row];
  int t   = threadIdx.x;
  float v[3];
#pragma unroll
  for (int e = 0; e < 3; ++e) {
    int c = t + e * 256;
    v[e] = wemb[(size_t)id * kH + c] + pemb[(size_t)s * kH + c];
  }
  float mean = blk_sum256(v[0] + v[1] + v[2]) * (1.0f / (float)kH);
  float d0 = v[0] - mean, d1 = v[1] - mean, d2 = v[2] - mean;
  float var = blk_sum256(d0 * d0 + d1 * d1 + d2 * d2) * (1.0f / (float)kH);
  float inv = rsqrtf(var + 1e-12f);
#pragma unroll
  for (int e = 0; e < 3; ++e) {
    int c = t + e * 256;
    float o = (v[e] - mean) * inv * g[c] + be[c];
    size_t idx = (size_t)row * kH + c;
    outF[idx] = o;
    outH[idx] = (_Float16)o;
  }
}

// ---------------- (x + resid) -> LayerNorm ----------------
__global__ __launch_bounds__(256) void addln_kernel(
    const float* __restrict__ xin, const float* __restrict__ resid,
    const float* __restrict__ g, const float* __restrict__ be,
    float* __restrict__ outF, _Float16* __restrict__ outH) {
  int row = blockIdx.x;
  int t   = threadIdx.x;
  float v[3];
#pragma unroll
  for (int e = 0; e < 3; ++e) {
    size_t idx = (size_t)row * kH + t + e * 256;
    v[e] = xin[idx] + (resid ? resid[idx] : 0.0f);
  }
  float mean = blk_sum256(v[0] + v[1] + v[2]) * (1.0f / (float)kH);
  float d0 = v[0] - mean, d1 = v[1] - mean, d2 = v[2] - mean;
  float var = blk_sum256(d0 * d0 + d1 * d1 + d2 * d2) * (1.0f / (float)kH);
  float inv = rsqrtf(var + 1e-12f);
#pragma unroll
  for (int e = 0; e < 3; ++e) {
    int c = t + e * 256;
    float o = (v[e] - mean) * inv * g[c] + be[c];
    size_t idx = (size_t)row * kH + c;
    if (outF) outF[idx] = o;
    if (outH) outH[idx] = (_Float16)o;
  }
}

// ---------------- masked softmax -> f16 probs ----------------
// mode 0: self (causal + answer padding); mode 1: cross (encoder padding)
__global__ __launch_bounds__(256) void softmax_kernel(
    const float* __restrict__ Sc, _Float16* __restrict__ P,
    const float* __restrict__ mask, int Sq, int Sk, int mode, float scale) {
  long long row = blockIdx.x;          // ((b*NH + h)*Sq + q)
  int q = (int)(row % Sq);
  long long r = row / Sq;
  int b = (int)(r / kNH);
  const float* srow = Sc + row * (long long)Sk;
  const float* mrow = mask + (long long)b * Sk;

  float mx = -3.4e38f;
  for (int j = threadIdx.x; j < Sk; j += 256) {
    float add;
    if (mode == 0) {
      float am = mrow[j];
      add = (((1.0f - am) + ((j > q) ? 1.0f : 0.0f)) > 0.0f) ? -10000.0f : 0.0f;
    } else {
      add = (1.0f - mrow[j]) * -10000.0f;
    }
    mx = fmaxf(mx, srow[j] * scale + add);
  }
  mx = blk_max256(mx);

  float sum = 0.0f;
  for (int j = threadIdx.x; j < Sk; j += 256) {
    float add;
    if (mode == 0) {
      float am = mrow[j];
      add = (((1.0f - am) + ((j > q) ? 1.0f : 0.0f)) > 0.0f) ? -10000.0f : 0.0f;
    } else {
      add = (1.0f - mrow[j]) * -10000.0f;
    }
    sum += __expf(srow[j] * scale + add - mx);
  }
  sum = blk_sum256(sum);
  float inv = 1.0f / sum;

  for (int j = threadIdx.x; j < Sk; j += 256) {
    float add;
    if (mode == 0) {
      float am = mrow[j];
      add = (((1.0f - am) + ((j > q) ? 1.0f : 0.0f)) > 0.0f) ? -10000.0f : 0.0f;
    } else {
      add = (1.0f - mrow[j]) * -10000.0f;
    }
    P[row * (long long)Sk + j] = (_Float16)(__expf(srow[j] * scale + add - mx) * inv);
  }
}

// ---------------- V transpose: [B*Sk, H] -> [B, NH, kVTD, Sk] (f16, padded) ----
// Rows d in [kDH, kVTD) are written as zeros so GEMM B-loads need no guard.
__global__ void transpose_v_kernel(const _Float16* __restrict__ vh,
                                   _Float16* __restrict__ vt, int Sk) {
  long long i = (long long)blockIdx.x * blockDim.x + threadIdx.x;
  long long total = (long long)kB * kNH * kVTD * Sk;
  if (i >= total) return;
  int skey = (int)(i % Sk);
  long long r = i / Sk;
  int d = (int)(r % kVTD); r /= kVTD;
  int h = (int)(r % kNH);
  int b = (int)(r / kNH);
  vt[i] = (d < kDH) ? vh[((size_t)b * Sk + skey) * kH + h * kDH + d]
                    : (_Float16)0.0f;
}

// ---------------- host orchestration ----------------
extern "C" void kernel_launch(void* const* d_in, const int* in_sizes, int n_in,
                              void* d_out, int out_size, void* d_ws, size_t ws_size,
                              hipStream_t stream) {
  (void)in_sizes; (void)n_in; (void)out_size; (void)ws_size;

  const int*   ids      = (const int*)d_in[0];
  const float* enc_out  = (const float*)d_in[1];
  const float* ans_mask = (const float*)d_in[2];
  const float* enc_mask = (const float*)d_in[3];

  // params: jax pytree leaves, dict keys sorted
  auto prm = [&](int i) -> const float* { return (const float*)d_in[4 + i]; };
  const int I_CLS_B = 0, I_CLS_BIAS = 1, I_CLS_LN_B = 2, I_CLS_LN_G = 3,
            I_CLS_W = 4, I_EMB_LN_B = 5, I_EMB_LN_G = 6;
  const int I_POS_EMB = 7 + 26 * kL, I_WORD_EMB = I_POS_EMB + 1;
  auto LB = [&](int l) { return 7 + 26 * l; };
  // layer leaves: enc{bk,bo,bq,bv,ln_b,ln_g,wk,wo,wq,wv}=+0..9,
  //               ffn{b1,b2,ln_b,ln_g,w1,w2}=+10..15, slf{...}=+16..25

  // ---- workspace layout ----
  char* base = (char*)d_ws;
  size_t off = 0;
  auto alloc = [&](size_t bytes) -> void* {
    void* p = base + off;
    off += (bytes + 255) & ~(size_t)255;
    return p;
  };
  float*    xF    = (float*)alloc((size_t)kMR * kH * 4);
  float*    aF    = (float*)alloc((size_t)kMR * kH * 4);
  float*    a2F   = (float*)alloc((size_t)kMR * kH * 4);
  float*    tmpF  = (float*)alloc((size_t)kMR * kH * 4);
  float*    scF   = (float*)alloc((size_t)kBN * kS * kSE * 4);
  _Float16* xh    = (_Float16*)alloc((size_t)kMR * kH * 2);
  _Float16* ah    = (_Float16*)alloc((size_t)kMR * kH * 2);
  _Float16* a2h   = (_Float16*)alloc((size_t)kMR * kH * 2);
  _Float16* qh    = (_Float16*)alloc((size_t)kMR * kH * 2);
  _Float16* kh    = (_Float16*)alloc((size_t)kMR * kH * 2);
  _Float16* vh    = (_Float16*)alloc((size_t)kMR * kH * 2);
  _Float16* ctxh  = (_Float16*)alloc((size_t)kMR * kH * 2);
  _Float16* th    = (_Float16*)alloc((size_t)kMR * kH * 2);
  _Float16* ench  = (_Float16*)alloc((size_t)kB * kSE * kH * 2);
  _Float16* Ph    = (_Float16*)alloc((size_t)kBN * kS * kSE * 2);
  _Float16* vtH   = (_Float16*)alloc((size_t)kBN * kVTD * kSE * 2);
  _Float16* ffnH  = (_Float16*)alloc((size_t)kMR * kFF * 2);
  _Float16* wembH = (_Float16*)alloc((size_t)kVpad * kH * 2);
  _Float16* wbuf  = (_Float16*)alloc((size_t)kH * kFF * 2);  // weight f16 scratch

  auto gemm = [&](const _Float16* A, const _Float16* Bw, const float* bias,
                  float* Cf, _Float16* Ch, int M, int N, int K, int lda, int ldb,
                  int ldc, int batch, int zDiv, long long aSb, long long aSh,
                  long long bSb, long long bSh, long long cSb, long long cSh,
                  int act, bool guard) {
    dim3 g((N + 127) / 128, M / 128, batch);
    if (guard)
      gemm_wmma_kernel<true><<<g, 256, 0, stream>>>(A, Bw, bias, Cf, Ch, M, N, K,
          lda, ldb, ldc, zDiv, aSb, aSh, bSb, bSh, cSb, cSh, act);
    else
      gemm_wmma_kernel<false><<<g, 256, 0, stream>>>(A, Bw, bias, Cf, Ch, M, N, K,
          lda, ldb, ldc, zDiv, aSb, aSh, bSb, bSh, cSb, cSh, act);
  };
  auto wconv = [&](const float* w, long long n) -> const _Float16* {
    convert_f16_kernel<<<(unsigned)((n + 255) / 256), 256, 0, stream>>>(w, wbuf, n);
    return wbuf;
  };

  const long long HH = (long long)kH * kH;
  const long long SH = (long long)kS * kH;       // per-batch A/B stride (rows*H)
  const long long SS = (long long)kS * kSE;      // per-head score block

  // tied embedding table -> f16 (pad rows zeroed for unguarded B loads)
  convert_f16_kernel<<<(unsigned)(((long long)kV * kH + 255) / 256), 256, 0, stream>>>(
      prm(I_WORD_EMB), wembH, (long long)kV * kH);
  zero_f16_kernel<<<(unsigned)(((long long)(kVpad - kV) * kH + 255) / 256), 256, 0,
                    stream>>>(wembH + (size_t)kV * kH, (long long)(kVpad - kV) * kH);
  convert_f16_kernel<<<(unsigned)(((long long)kB * kSE * kH + 255) / 256), 256, 0, stream>>>(
      enc_out, ench, (long long)kB * kSE * kH);

  // embeddings + LN
  embed_ln_kernel<<<kMR, 256, 0, stream>>>(ids, prm(I_WORD_EMB), prm(I_POS_EMB),
                                           prm(I_EMB_LN_G), prm(I_EMB_LN_B), xF, xh);

  const long long vtSh = (long long)kVTD * kSE;          // per-head V^T stride
  const long long vtSb = (long long)kNH * kVTD * kSE;    // per-batch V^T stride

  for (int l = 0; l < kL; ++l) {
    int lb = LB(l);
    // ---------- self attention ----------
    int s_bk = lb + 16, s_bo = lb + 17, s_bq = lb + 18, s_bv = lb + 19,
        s_lnb = lb + 20, s_lng = lb + 21, s_wk = lb + 22, s_wo = lb + 23,
        s_wq = lb + 24, s_wv = lb + 25;
    gemm(xh, wconv(prm(s_wq), HH), prm(s_bq), nullptr, qh,
         kMR, kH, kH, kH, kH, kH, 1, 1, 0, 0, 0, 0, 0, 0, 0, false);
    gemm(xh, wconv(prm(s_wk), HH), prm(s_bk), nullptr, kh,
         kMR, kH, kH, kH, kH, kH, 1, 1, 0, 0, 0, 0, 0, 0, 0, false);
    gemm(xh, wconv(prm(s_wv), HH), prm(s_bv), nullptr, vh,
         kMR, kH, kH, kH, kH, kH, 1, 1, 0, 0, 0, 0, 0, 0, 0, false);
    gemm(qh, kh, nullptr, scF, nullptr, kS, kSE, kDH, kH, kH, kSE,
         kBN, kNH, SH, kDH, SH, kDH, (long long)kNH * SS, SS, 0, false);
    softmax_kernel<<<kBN * kS, 256, 0, stream>>>(scF, Ph, ans_mask, kS, kSE, 0, 0.125f);
    transpose_v_kernel<<<(unsigned)(((long long)kBN * kVTD * kSE + 255) / 256), 256, 0,
                         stream>>>(vh, vtH, kSE);
    gemm(Ph, vtH, nullptr, nullptr, ctxh, kS, kDH, kSE, kSE, kSE, kH,
         kBN, kNH, (long long)kNH * SS, SS, vtSb, vtSh, SH, kDH, 0, true);
    gemm(ctxh, wconv(prm(s_wo), HH), prm(s_bo), tmpF, nullptr,
         kMR, kH, kH, kH, kH, kH, 1, 1, 0, 0, 0, 0, 0, 0, 0, false);
    addln_kernel<<<kMR, 256, 0, stream>>>(tmpF, xF, prm(s_lng), prm(s_lnb), aF, ah);

    // ---------- cross attention ----------
    int e_bk = lb + 0, e_bo = lb + 1, e_bq = lb + 2, e_bv = lb + 3,
        e_lnb = lb + 4, e_lng = lb + 5, e_wk = lb + 6, e_wo = lb + 7,
        e_wq = lb + 8, e_wv = lb + 9;
    gemm(ah, wconv(prm(e_wq), HH), prm(e_bq), nullptr, qh,
         kMR, kH, kH, kH, kH, kH, 1, 1, 0, 0, 0, 0, 0, 0, 0, false);
    gemm(ench, wconv(prm(e_wk), HH), prm(e_bk), nullptr, kh,
         kB * kSE, kH, kH, kH, kH, kH, 1, 1, 0, 0, 0, 0, 0, 0, 0, false);
    gemm(ench, wconv(prm(e_wv), HH), prm(e_bv), nullptr, vh,
         kB * kSE, kH, kH, kH, kH, kH, 1, 1, 0, 0, 0, 0, 0, 0, 0, false);
    gemm(qh, kh, nullptr, scF, nullptr, kS, kSE, kDH, kH, kH, kSE,
         kBN, kNH, SH, kDH, (long long)kSE * kH, kDH, (long long)kNH * SS, SS, 0, false);
    softmax_kernel<<<kBN * kS, 256, 0, stream>>>(scF, Ph, enc_mask, kS, kSE, 1, 0.125f);
    transpose_v_kernel<<<(unsigned)(((long long)kBN * kVTD * kSE + 255) / 256), 256, 0,
                         stream>>>(vh, vtH, kSE);
    gemm(Ph, vtH, nullptr, nullptr, ctxh, kS, kDH, kSE, kSE, kSE, kH,
         kBN, kNH, (long long)kNH * SS, SS, vtSb, vtSh, SH, kDH, 0, true);
    gemm(ctxh, wconv(prm(e_wo), HH), prm(e_bo), tmpF, nullptr,
         kMR, kH, kH, kH, kH, kH, 1, 1, 0, 0, 0, 0, 0, 0, 0, false);
    addln_kernel<<<kMR, 256, 0, stream>>>(tmpF, aF, prm(e_lng), prm(e_lnb), a2F, a2h);

    // ---------- feed-forward ----------
    gemm(a2h, wconv(prm(lb + 14), (long long)kFF * kH), prm(lb + 10), nullptr, ffnH,
         kMR, kFF, kH, kH, kH, kFF, 1, 1, 0, 0, 0, 0, 0, 0, /*gelu=*/1, false);
    gemm(ffnH, wconv(prm(lb + 15), (long long)kH * kFF), prm(lb + 11), tmpF, nullptr,
         kMR, kH, kFF, kFF, kFF, kH, 1, 1, 0, 0, 0, 0, 0, 0, 0, false);
    addln_kernel<<<kMR, 256, 0, stream>>>(tmpF, a2F, prm(lb + 13), prm(lb + 12), xF, xh);
  }

  // ---------- classifier head ----------
  gemm(xh, wconv(prm(I_CLS_W), HH), prm(I_CLS_B), tmpF, nullptr,
       kMR, kH, kH, kH, kH, kH, 1, 1, 0, 0, 0, 0, 0, 0, /*gelu=*/1, false);
  addln_kernel<<<kMR, 256, 0, stream>>>(tmpF, nullptr, prm(I_CLS_LN_G),
                                        prm(I_CLS_LN_B), nullptr, th);
  gemm(th, wembH, prm(I_CLS_BIAS), (float*)d_out, nullptr,
       kMR, kV, kH, kH, kH, kV, 1, 1, 0, 0, 0, 0, 0, 0, 0, true);
}
